// DecoderOnlyTransformer_49168785604683
// MI455X (gfx1250) — matrix operand order repaired and checked by reading, not compiled, not run on hardware
//
#include <hip/hip_runtime.h>
#include <hip/hip_bf16.h>

// ---------------------------------------------------------------------------
// Types for CDNA5 WMMA (wave32): v_wmma_f32_16x16x32_bf16
// ---------------------------------------------------------------------------
typedef __attribute__((ext_vector_type(16))) __bf16 bf16x16;
typedef __attribute__((ext_vector_type(8)))  float  floatx8;

union ABFrag {
  unsigned int u[8];
  bf16x16 v;
};

__device__ __forceinline__ __bf16 f2bf(float f) {
  __hip_bfloat16 b = __float2bfloat16(f);
  __bf16 r;
  __builtin_memcpy(&r, &b, sizeof(r));
  return r;
}

// CDNA5 async global->LDS DMA (ASYNCcnt). dsaddr = lds byte address (low 32
// bits of the generic pointer = LDS aperture offset); both addresses get the
// same immediate offset per the ISA pseudocode.
__device__ __forceinline__ void async_copy_b128(unsigned lds, unsigned long long ga) {
  asm volatile("global_load_async_to_lds_b128 %0, %1, off"
               :: "v"(lds), "v"(ga) : "memory");
}
__device__ __forceinline__ void async_copy_b128_off16(unsigned lds, unsigned long long ga) {
  asm volatile("global_load_async_to_lds_b128 %0, %1, off offset:16"
               :: "v"(lds), "v"(ga) : "memory");
}
__device__ __forceinline__ void wait_asynccnt0() {
  asm volatile("s_wait_asynccnt 0x0" ::: "memory");
}

// ---------------------------------------------------------------------------
// Embedding gather + positional encoding -> bf16 activations
// ---------------------------------------------------------------------------
__global__ __launch_bounds__(256) void embed_pe_kernel(
    const int* __restrict__ x, const float* __restrict__ emb,
    const float* __restrict__ pe, __bf16* __restrict__ h,
    int S, int D)
{
  int bs  = blockIdx.x;         // b*S + s
  int s   = bs % S;
  int tok = x[bs];
  const float* e = emb + (size_t)tok * D;
  const float* p = pe  + (size_t)s   * D;
  __bf16* out = h + (size_t)bs * D;
  for (int d = threadIdx.x; d < D; d += 256)
    out[d] = f2bf(e[d] + p[d]);
}

// ---------------------------------------------------------------------------
// Causal softmax over fp32 scores -> bf16 probs. One block per (b, query row).
// S must equal 2048 (8 elements per thread at 256 threads).
// ---------------------------------------------------------------------------
__global__ __launch_bounds__(256) void softmax_causal_kernel(
    const float* __restrict__ scores, __bf16* __restrict__ probs,
    int S, float scale)
{
  size_t row = blockIdx.x;              // b*S + q
  int qi = (int)(row % (size_t)S);
  const float* src = scores + row * (size_t)S;
  __bf16*      dst = probs  + row * (size_t)S;

  __shared__ float red[8];
  int t = threadIdx.x;
  int lane = t & 31, wid = t >> 5;
  const float NEG_INF = -__builtin_inff();

  float vals[8];
  float mx = NEG_INF;
#pragma unroll
  for (int i = 0; i < 8; i++) {
    int j = t + i * 256;
    float v = (j <= qi) ? src[j] * scale : NEG_INF;
    vals[i] = v;
    mx = fmaxf(mx, v);
  }
  for (int o = 16; o > 0; o >>= 1) mx = fmaxf(mx, __shfl_down(mx, o));
  if (lane == 0) red[wid] = mx;
  __syncthreads();
  if (t == 0) {
    float m = red[0];
    for (int i = 1; i < 8; i++) m = fmaxf(m, red[i]);
    red[0] = m;
  }
  __syncthreads();
  mx = red[0];

  float sum = 0.f;
#pragma unroll
  for (int i = 0; i < 8; i++) {
    float e = (vals[i] == NEG_INF) ? 0.f : __expf(vals[i] - mx);
    vals[i] = e;
    sum += e;
  }
  for (int o = 16; o > 0; o >>= 1) sum += __shfl_down(sum, o);
  __syncthreads();                       // everyone done reading red[0]
  if (lane == 0) red[wid] = sum;
  __syncthreads();
  if (t == 0) {
    float s2 = 0.f;
    for (int i = 0; i < 8; i++) s2 += red[i];
    red[0] = s2;
  }
  __syncthreads();
  float inv = 1.f / red[0];
#pragma unroll
  for (int i = 0; i < 8; i++)
    dst[t + i * 256] = f2bf(vals[i] * inv);
}

// ---------------------------------------------------------------------------
// Tiled bf16 GEMM via v_wmma_f32_16x16x32_bf16, fp32 accumulate.
//   C[M,N] = A[M,K](bf16,row-major) x B[K,N] (+ bias) (+ gelu)
// Block tile 128x128x32; 8 wave32s (2x4), each wave 64x32 = 4x2 WMMA tiles.
// BSRC: 0 = fp32 B[k][n] row-major (b128 loads, convert->bf16, transpose)
//       1 = bf16 X[n][k] row-major, B = X^T (async DMA copy into LDS)
//       2 = bf16 B[k][n] row-major (b128 loads, transpose into LDS)
// OUT_F32: 1 -> fp32 C, else bf16 C.  GELU_EPI: exact gelu on result.
// CAUSAL: skip blocks entirely above the diagonal (scores GEMM).
// M,N multiples of 128; K multiple of 32. blockIdx.z batches via strides.
// ---------------------------------------------------------------------------
#define TILE_K 32
#define LDS_STRIDE 40   // padded row stride in elements (80B: 16B aligned, conflict-free)

template<int BSRC, int OUT_F32, int GELU_EPI, int CAUSAL>
__global__ __launch_bounds__(256) void gemm_bf16_wmma(
    const __bf16* __restrict__ A, long long lda, long long strideA,
    const void*  __restrict__ Bsrc, long long ldb, long long strideB,
    const float* __restrict__ bias,
    void* __restrict__ Cout, long long ldc, long long strideC,
    int M, int N, int K)
{
  const int bm = blockIdx.y;
  const int bn = blockIdx.x;
  if (CAUSAL && bn > bm) return;        // fully masked key block
  const long long bz = blockIdx.z;

  const __bf16* Ab = A + bz * strideA;
  const __bf16* Bb = (const __bf16*)Bsrc + bz * strideB;
  const float*  Bf = (const float*)Bsrc + bz * strideB;

  __shared__ __bf16 sA[128 * LDS_STRIDE];   // A tile, row-major [m][k]
  __shared__ __bf16 sB[128 * LDS_STRIDE];   // B tile, transposed [n][k]

  const int t    = threadIdx.x;
  const int wid  = t >> 5;
  const int lane = t & 31;
  const int wm   = wid & 1;     // 0..1  (64 rows each)
  const int wn   = wid >> 1;    // 0..3  (32 cols each)
  const int lr   = lane & 15;
  const int hi   = lane >> 4;   // 0 or 1

  floatx8 acc[4][2];
#pragma unroll
  for (int i = 0; i < 4; i++)
#pragma unroll
    for (int j = 0; j < 2; j++) acc[i][j] = {};

  for (int k0 = 0; k0 < K; k0 += TILE_K) {
    __syncthreads();
    // ---- stage A: 128 rows x 32 bf16 via async DMA (2 x b128 per thread) ----
    {
      int row = t >> 1;
      int c0  = (t & 1) << 4;
      const __bf16* gsrc = Ab + ((size_t)(bm * 128 + row)) * lda + k0 + c0;
      unsigned lds =
          (unsigned)(unsigned long long)(const void*)(sA + row * LDS_STRIDE + c0);
      unsigned long long ga = (unsigned long long)gsrc;
      async_copy_b128(lds, ga);
      async_copy_b128_off16(lds, ga);
    }
    // ---- stage B into transposed LDS layout Bt[n][k] ----
    if (BSRC == 0) {               // fp32 W[k][n]: b128 loads, convert + transpose
      int kk = t >> 3;
      int n0 = (t & 7) << 4;
      const float4* src4 =
          (const float4*)(Bf + (size_t)(k0 + kk) * ldb + (size_t)bn * 128 + n0);
      union { float4 q[4]; float f[16]; } wbuf;
      wbuf.q[0] = src4[0];
      wbuf.q[1] = src4[1];
      wbuf.q[2] = src4[2];
      wbuf.q[3] = src4[3];
      if (k0 + TILE_K < K)         // stream next K-tile of the weight matrix
        __builtin_prefetch((const void*)(src4 + (size_t)TILE_K * ldb / 4), 0, 1);
#pragma unroll
      for (int i = 0; i < 16; i++)
        sB[(n0 + i) * LDS_STRIDE + kk] = f2bf(wbuf.f[i]);
    } else if (BSRC == 1) {        // bf16 X[n][k], B = X^T: async DMA copy
      int n  = t >> 1;
      int c0 = (t & 1) << 4;
      const __bf16* gsrc = Bb + ((size_t)(bn * 128 + n)) * ldb + k0 + c0;
      unsigned lds =
          (unsigned)(unsigned long long)(const void*)(sB + n * LDS_STRIDE + c0);
      unsigned long long ga = (unsigned long long)gsrc;
      async_copy_b128(lds, ga);
      async_copy_b128_off16(lds, ga);
    } else {                       // bf16 B[k][n]: b128 loads, transpose copy
      int kk = t >> 3;
      int n0 = (t & 7) << 4;
      const uint4* s4 =
          (const uint4*)(Bb + (size_t)(k0 + kk) * ldb + (size_t)bn * 128 + n0);
      union { uint4 q[2]; __bf16 e[16]; } vb;
      vb.q[0] = s4[0];
      vb.q[1] = s4[1];
#pragma unroll
      for (int i = 0; i < 16; i++)
        sB[(n0 + i) * LDS_STRIDE + kk] = vb.e[i];
    }
    wait_asynccnt0();              // async DMA landed in LDS
    __syncthreads();

    // ---- load fragments, A and B both in contiguous k order (k = 16*hi + s) ----
    ABFrag afr[4], bfr[2];
#pragma unroll
    for (int mt = 0; mt < 4; mt++) {
      int row = wm * 64 + mt * 16 + lr;
      const unsigned int* p = (const unsigned int*)(sA + row * LDS_STRIDE);
      int kb = hi ? 8 : 0;          // u32 units (lane>=16 -> K base 16)
#pragma unroll
      for (int j = 0; j < 4; j++) afr[mt].u[j] = p[kb + j];
#pragma unroll
      for (int j = 0; j < 4; j++) afr[mt].u[4 + j] = p[kb + 4 + j];
    }
#pragma unroll
    for (int nt = 0; nt < 2; nt++) {
      int col = wn * 32 + nt * 16 + lr;
      const unsigned int* p = (const unsigned int*)(sB + col * LDS_STRIDE);
      int kb = hi ? 8 : 0;          // u32 units (lane>=16 -> K base 16)
#pragma unroll
      for (int j = 0; j < 8; j++) bfr[nt].u[j] = p[kb + j];
    }

#pragma unroll
    for (int mt = 0; mt < 4; mt++)
#pragma unroll
      for (int nt = 0; nt < 2; nt++)
        acc[mt][nt] = __builtin_amdgcn_wmma_f32_16x16x32_bf16(
            false, afr[mt].v, false, bfr[nt].v, (short)0, acc[mt][nt],
            false, false);
  }

  // ---- epilogue: bias (+gelu) and store per documented C layout ----
#pragma unroll
  for (int mt = 0; mt < 4; mt++) {
#pragma unroll
    for (int nt = 0; nt < 2; nt++) {
      int gcol = bn * 128 + wn * 32 + nt * 16 + lr;
      float bv = bias ? bias[gcol] : 0.f;
#pragma unroll
      for (int r = 0; r < 8; r++) {
        int grow = bm * 128 + wm * 64 + mt * 16 + (hi ? 8 : 0) + r;
        float v = acc[mt][nt][r] + bv;
        if (GELU_EPI)
          v = 0.5f * v * (1.f + erff(v * 0.70710678118654752440f));
        size_t idx = (size_t)bz * (size_t)strideC + (size_t)grow * ldc + gcol;
        if (OUT_F32) ((float*)Cout)[idx] = v;
        else         ((__bf16*)Cout)[idx] = f2bf(v);
      }
    }
  }
}

// ---------------------------------------------------------------------------
// Launch: embed -> QKV -> causal scores -> softmax -> probs@V -> O -> FF1
//         (gelu) -> vocab projection (fp32 out).
// ---------------------------------------------------------------------------
extern "C" void kernel_launch(void* const* d_in, const int* in_sizes, int n_in,
                              void* d_out, int out_size, void* d_ws, size_t ws_size,
                              hipStream_t stream)
{
  (void)in_sizes; (void)n_in; (void)out_size; (void)ws_size;

  const int B = 2, S = 2048, D = 1024, Mf = 4096, OMEGA = 32000;
  const int N = B * S;   // 4096 token rows

  const int*   x   = (const int*)  d_in[0];
  const float* emb = (const float*)d_in[1];
  const float* pe  = (const float*)d_in[2];
  const float* Wq  = (const float*)d_in[3];
  const float* bq  = (const float*)d_in[4];
  const float* Wk  = (const float*)d_in[5];
  const float* bk  = (const float*)d_in[6];
  const float* Wv  = (const float*)d_in[7];
  const float* bv  = (const float*)d_in[8];
  const float* Wo  = (const float*)d_in[9];
  const float* bo  = (const float*)d_in[10];
  const float* Wf  = (const float*)d_in[11];
  const float* bf_ = (const float*)d_in[12];
  const float* Wl  = (const float*)d_in[13];
  const float* bl  = (const float*)d_in[14];

  // Workspace layout (128 MB total)
  char* ws = (char*)d_ws;
  __bf16* h_bf    = (__bf16*)(ws);                       //  8 MB  [N,D]
  __bf16* q_bf    = (__bf16*)(ws + (size_t)8  * (1u<<20));
  __bf16* k_bf    = (__bf16*)(ws + (size_t)16 * (1u<<20));
  __bf16* v_bf    = (__bf16*)(ws + (size_t)24 * (1u<<20));
  __bf16* attn_bf = (__bf16*)(ws + (size_t)32 * (1u<<20));
  __bf16* ao_bf   = (__bf16*)(ws + (size_t)40 * (1u<<20));
  __bf16* ff_bf   = (__bf16*)(ws + (size_t)48 * (1u<<20)); // 32 MB [N,Mf]
  float*  sc      = (float*) (ws + (size_t)80 * (1u<<20)); // 32 MB [B,S,S]
  __bf16* pr      = (__bf16*)(ws + (size_t)112* (1u<<20)); // 16 MB [B,S,S]

  dim3 blk(256);

  // 1) h = emb[x] + pe  (bf16)
  embed_pe_kernel<<<N, 256, 0, stream>>>(x, emb, pe, h_bf, S, D);

  // 2) Q, K, V projections: [N,D] x [D,D] + bias
  gemm_bf16_wmma<0,0,0,0><<<dim3(D/128, N/128, 1), blk, 0, stream>>>(
      h_bf, D, 0, Wq, D, 0, bq, q_bf, D, 0, N, D, D);
  gemm_bf16_wmma<0,0,0,0><<<dim3(D/128, N/128, 1), blk, 0, stream>>>(
      h_bf, D, 0, Wk, D, 0, bk, k_bf, D, 0, N, D, D);
  gemm_bf16_wmma<0,0,0,0><<<dim3(D/128, N/128, 1), blk, 0, stream>>>(
      h_bf, D, 0, Wv, D, 0, bv, v_bf, D, 0, N, D, D);

  // 3) scores[b] = q[b] @ k[b]^T  (fp32, causal block skip; scale in softmax)
  gemm_bf16_wmma<1,1,0,1><<<dim3(S/128, S/128, B), blk, 0, stream>>>(
      q_bf, D, (long long)S * D, k_bf, D, (long long)S * D, nullptr,
      sc, S, (long long)S * S, S, S, D);

  // 4) causal softmax with 1/sqrt(D) scaling -> bf16 probs
  softmax_causal_kernel<<<B * S, 256, 0, stream>>>(sc, pr, S, 1.0f / 32.0f);

  // 5) attn[b] = probs[b] @ v[b]   [S,S] x [S,D]
  gemm_bf16_wmma<2,0,0,0><<<dim3(D/128, S/128, B), blk, 0, stream>>>(
      pr, S, (long long)S * S, v_bf, D, (long long)S * D, nullptr,
      attn_bf, D, (long long)S * D, S, D, S);

  // 6) O projection: [N,D] x [D,D] + bias
  gemm_bf16_wmma<0,0,0,0><<<dim3(D/128, N/128, 1), blk, 0, stream>>>(
      attn_bf, D, 0, Wo, D, 0, bo, ao_bf, D, 0, N, D, D);

  // 7) FF1 + exact GELU: [N,D] x [D,Mf] + bias
  gemm_bf16_wmma<0,0,1,0><<<dim3(Mf/128, N/128, 1), blk, 0, stream>>>(
      ao_bf, D, 0, Wf, Mf, 0, bf_, ff_bf, Mf, 0, N, Mf, D);

  // 8) vocab projection (fp32 out): [N,Mf] x [Mf,OMEGA] + bias
  gemm_bf16_wmma<0,1,0,0><<<dim3(OMEGA/128, N/128, 1), blk, 0, stream>>>(
      ff_bf, Mf, 0, Wl, OMEGA, 0, bl, (float*)d_out, OMEGA, 0, N, OMEGA, Mf);
}
